// cross_attention_19963007992233
// MI455X (gfx1250) — matrix operand order repaired and checked by reading, not compiled, not run on hardware
//
#include <hip/hip_runtime.h>
#include <hip/hip_bf16.h>

// MI455X / gfx1250 axial-attention block.
// All GEMMs use v_wmma_f32_16x16x32_bf16 (bf16 inputs, f32 accumulate).
// Staging copies of bf16 tiles use global_load_async_to_lds_b128 (ASYNCcnt).
// Workspace layout (bf16, NT = 16*128*128*128 elements each):
//   [0]=theta(pm) [1]=phi(pm) [2]=g(cm) [3]=sc_h(cm) [4]=sc_v(cm) [5]=f_h(pm) [6]=f_v(pm)
// pm = pixel-major [b,h,w,c], cm = channel-major [b,c,h,w]. Needs ~470 MB scratch.

typedef __bf16 bf16;
typedef __bf16 bf16x8  __attribute__((ext_vector_type(8)));
typedef __bf16 bf16x16 __attribute__((ext_vector_type(16)));
typedef float  f32x4   __attribute__((ext_vector_type(4)));
typedef float  f32x8   __attribute__((ext_vector_type(8)));

#define DI __device__ __forceinline__

constexpr int        HW  = 128 * 128;                 // 16384
constexpr long long  NT  = 16LL * 128 * 128 * 128;    // 33,554,432 elems per tensor
constexpr int        PT  = 136;                       // bf16 LDS tile pitch (272B rows, 16B aligned)
constexpr int        SP  = 129;                       // f32 score pitch (bank spread)
constexpr float      INV_SQRT_D = 0.08838834764831845f;

constexpr unsigned SMEM_PROJ = 128u * PT * 2 + 16384u * 2;                          // 67,584 B
constexpr unsigned SMEM_ATTN = 3u * 128 * PT * 2 + 128u * SP * 4 + 128u * PT * 2;   // 205,312 B
constexpr unsigned SMEM_FUSE = 2u * 128 * PT * 2;                                   // 69,632 B

// ---- CDNA5 async global->LDS copy (ASYNCcnt) -------------------------------
// One 16-byte transfer per lane: LDS byte address in a VGPR (low 32 bits of the
// flat LDS-aperture address == LDS offset per ISA 10.2), 64-bit global VA pair.
DI void async_ld_b128(const void* gptr, void* lptr) {
  unsigned lds_addr = (unsigned)(unsigned long long)lptr;
  unsigned long long ga = (unsigned long long)gptr;
  asm volatile("global_load_async_to_lds_b128 %0, %1, off"
               :: "v"(lds_addr), "v"(ga)
               : "memory");
}
DI void wait_async0() { asm volatile("s_wait_asynccnt 0x0" ::: "memory"); }

// ---- WMMA fragment helpers -------------------------------------------------
// A fragment (16x32 bf16): lane row m, halves {c0..c0+7, c0+16..c0+23}
DI bf16x16 ldsA_frag(const bf16* row, int c0) {
  bf16x8 lo = *(const bf16x8*)(row + c0);
  bf16x8 hi = *(const bf16x8*)(row + c0 + 16);
  bf16x16 r;
#pragma unroll
  for (int j = 0; j < 8; ++j) { r[j] = lo[j]; r[j + 8] = hi[j]; }
  return r;
}
// B fragment (32x16 bf16): lane column n, halves {c0..c0+7, c0+16..c0+23}
DI bf16x16 ldsB_frag(const bf16* row, int c0) {
  bf16x8 lo = *(const bf16x8*)(row + c0);
  bf16x8 hi = *(const bf16x8*)(row + c0 + 16);
  bf16x16 r;
#pragma unroll
  for (int j = 0; j < 8; ++j) { r[j] = lo[j]; r[j + 8] = hi[j]; }
  return r;
}
// A fragment straight from fp32 weights in global memory (converted to bf16)
DI bf16x16 gldA_f32(const float* row, int c0) {
  f32x4 a = *(const f32x4*)(row + c0);
  f32x4 b = *(const f32x4*)(row + c0 + 4);
  f32x4 c = *(const f32x4*)(row + c0 + 16);
  f32x4 d = *(const f32x4*)(row + c0 + 20);
  bf16x16 r;
#pragma unroll
  for (int j = 0; j < 4; ++j) {
    r[j]      = (bf16)a[j];
    r[4 + j]  = (bf16)b[j];
    r[8 + j]  = (bf16)c[j];
    r[12 + j] = (bf16)d[j];
  }
  return r;
}
DI f32x8 wmma_bf16(bf16x16 a, bf16x16 b, f32x8 c) {
  return __builtin_amdgcn_wmma_f32_16x16x32_bf16(false, a, false, b, (short)0, c,
                                                 false, false);
}

// ---- Kernel 1: projections + shortcut convs --------------------------------
// Per (b,h) row-tile: X^T staged in LDS, 5 GEMMs: theta, phi, g, sc_h, sc_v.
__global__ void __launch_bounds__(256) proj_kernel(
    const float* __restrict__ x, const float* __restrict__ W7,
    const float* __restrict__ scale, const float* __restrict__ shift,
    bf16* __restrict__ theta, bf16* __restrict__ phi, bf16* __restrict__ g,
    bf16* __restrict__ sch, bf16* __restrict__ scv) {
  extern __shared__ char smem[];
  bf16* XT = (bf16*)smem;                    // [128 w][PT c]
  bf16* SO = (bf16*)(smem + 128 * PT * 2);   // [16384] staging

  const int b = blockIdx.x >> 7, h = blockIdx.x & 127;
  const int tid = threadIdx.x, lane = tid & 31, wave = tid >> 5;
  const int hi16 = (lane >> 4) & 1, l16 = lane & 15;

  const float* xbh = x + ((long long)b * 128) * HW + h * 128;  // x[b,c,h,w]=xbh[c*HW+w]
  for (int idx = tid; idx < 16384; idx += 256) {
    int c = idx >> 7, w = idx & 127;
    XT[w * PT + c] = (bf16)xbh[(long long)c * HW + w];
  }
  __syncthreads();

  const int  widx[5] = {0, 1, 2, 4, 6};
  const int  relu_f[5] = {1, 1, 1, 0, 0};
  const int  pm_f[5] = {1, 1, 0, 0, 0};
  bf16* outp[5] = {theta, phi, g, sch, scv};

#pragma unroll
  for (int i = 0; i < 5; ++i) {
    const float* Wrow = W7 + widx[i] * 16384 + (wave * 16 + l16) * 128;
    const int c0a = hi16 ? 8 : 0;
    bf16x16 a0 = gldA_f32(Wrow, 0 + c0a);
    bf16x16 a1 = gldA_f32(Wrow, 32 + c0a);
    bf16x16 a2 = gldA_f32(Wrow, 64 + c0a);
    bf16x16 a3 = gldA_f32(Wrow, 96 + c0a);
    float sc8[8], sh8[8];
#pragma unroll
    for (int rr = 0; rr < 8; ++rr) {
      int o = wave * 16 + rr + hi16 * 8;
      sc8[rr] = scale[widx[i] * 128 + o];
      sh8[rr] = shift[widx[i] * 128 + o];
    }
#pragma unroll
    for (int nt = 0; nt < 8; ++nt) {
      f32x8 acc = {};
      const bf16* brow = XT + (nt * 16 + l16) * PT;
      const int cb = hi16 ? 8 : 0;
      acc = wmma_bf16(a0, ldsB_frag(brow, 0 + cb), acc);
      acc = wmma_bf16(a1, ldsB_frag(brow, 32 + cb), acc);
      acc = wmma_bf16(a2, ldsB_frag(brow, 64 + cb), acc);
      acc = wmma_bf16(a3, ldsB_frag(brow, 96 + cb), acc);
      const int w = nt * 16 + l16;
#pragma unroll
      for (int rr = 0; rr < 8; ++rr) {
        int o = wave * 16 + rr + hi16 * 8;
        float v = acc[rr] * sc8[rr] + sh8[rr];
        if (relu_f[i]) v = fmaxf(v, 0.f);
        if (pm_f[i]) SO[w * 128 + o] = (bf16)v;
        else         SO[o * 128 + w] = (bf16)v;
      }
    }
    __syncthreads();
    bf16* dst = outp[i];
    const unsigned* s32 = (const unsigned*)SO;
    if (pm_f[i]) {
      unsigned* d32 = (unsigned*)(dst + (((long long)b * 128 + h) * 128) * 128);
      for (int idx = tid; idx < 8192; idx += 256) d32[idx] = s32[idx];
    } else {
      for (int idx = tid; idx < 8192; idx += 256) {
        int o = idx >> 6, w2 = idx & 63;
        *(unsigned*)(dst + (((long long)b * 128 + o) * 128 + h) * 128 + w2 * 2) = s32[idx];
      }
    }
    __syncthreads();
  }
}

// ---- Kernels 2/3: axial attention (VERT=0 horizontal rows, VERT=1 columns) --
template <int VERT>
__global__ void __launch_bounds__(256) attn_kernel(
    const float* __restrict__ W7, const float* __restrict__ scale,
    const float* __restrict__ shift, const bf16* __restrict__ theta,
    const bf16* __restrict__ phi, const bf16* __restrict__ g,
    const bf16* __restrict__ sc_in, bf16* __restrict__ fout) {
  extern __shared__ char smem[];
  bf16*  TH = (bf16*)smem;                              // [q][c]
  bf16*  PH = (bf16*)(smem + 128 * PT * 2);             // [k][c]
  bf16*  G  = (bf16*)(smem + 2 * 128 * PT * 2);         // [c][k]
  float* S  = (float*)(smem + 3 * 128 * PT * 2);        // [q][SP]
  bf16*  A  = (bf16*)(smem + 3 * 128 * PT * 2 + 128 * SP * 4);  // [q][v]
  bf16*  OT = TH;   // reuse after S-GEMM: O^T [q][c]
  bf16*  FH = PH;   // reuse after S-GEMM: f    [q][o] (pitch 128)

  const int b = blockIdx.x >> 7, r = blockIdx.x & 127;
  const int tid = threadIdx.x, lane = tid & 31, wave = tid >> 5;
  const int hi16 = (lane >> 4) & 1, l16 = lane & 15;

  // Stage theta/phi rows (128 rows x 256B) via async copies: 16 x b128 per row.
  for (int idx = tid; idx < 2048; idx += 256) {
    int q = idx >> 4, seg = (idx & 15) * 8;  // seg in halfs
    long long srow = VERT ? ((((long long)b * 128 + q) * 128 + r) * 128)
                          : ((((long long)b * 128 + r) * 128 + q) * 128);
    async_ld_b128(theta + srow + seg, TH + q * PT + seg);
    async_ld_b128(phi + srow + seg, PH + q * PT + seg);
  }
  if (!VERT) {
    // g rows contiguous for horizontal pass: async as well
    for (int idx = tid; idx < 2048; idx += 256) {
      int c = idx >> 4, seg = (idx & 15) * 8;
      async_ld_b128(g + (((long long)b * 128 + c) * 128 + r) * 128 + seg,
                    G + c * PT + seg);
    }
  } else {
    // vertical: strided gather (stride 128 elems) stays on the VGPR path
    for (int idx = tid; idx < 16384; idx += 256) {
      int c = idx >> 7, k = idx & 127;
      G[c * PT + k] = g[(((long long)b * 128 + c) * 128 + k) * 128 + r];
    }
  }
  wait_async0();
  __syncthreads();

  {  // S[q][k] = (1/sqrt(d)) * sum_c theta[c,q]*phi[c,k]
    const bf16* arow = TH + (wave * 16 + l16) * PT;
    const int c0a = hi16 ? 8 : 0, cb = hi16 ? 8 : 0;
    bf16x16 a0 = ldsA_frag(arow, 0 + c0a), a1 = ldsA_frag(arow, 32 + c0a);
    bf16x16 a2 = ldsA_frag(arow, 64 + c0a), a3 = ldsA_frag(arow, 96 + c0a);
#pragma unroll
    for (int nt = 0; nt < 8; ++nt) {
      f32x8 acc = {};
      const bf16* brow = PH + (nt * 16 + l16) * PT;
      acc = wmma_bf16(a0, ldsB_frag(brow, 0 + cb), acc);
      acc = wmma_bf16(a1, ldsB_frag(brow, 32 + cb), acc);
      acc = wmma_bf16(a2, ldsB_frag(brow, 64 + cb), acc);
      acc = wmma_bf16(a3, ldsB_frag(brow, 96 + cb), acc);
#pragma unroll
      for (int rr = 0; rr < 8; ++rr) {
        int q = wave * 16 + rr + hi16 * 8, k = nt * 16 + l16;
        S[q * SP + k] = acc[rr] * INV_SQRT_D;
      }
    }
  }
  __syncthreads();
  if (tid < 128) {  // softmax over keys, emit bf16 attention row
    float m = -3.4e38f;
    for (int v = 0; v < 128; ++v) m = fmaxf(m, S[tid * SP + v]);
    float s = 0.f;
    for (int v = 0; v < 128; ++v) {
      float e = __expf(S[tid * SP + v] - m);
      S[tid * SP + v] = e;
      s += e;
    }
    float inv = 1.f / s;
    for (int v = 0; v < 128; ++v) A[tid * PT + v] = (bf16)(S[tid * SP + v] * inv);
  }
  __syncthreads();

  {  // O[c][q] = sum_v G[c][v]*A[q][v]  -> store O^T [q][c]
    const bf16* arow = G + (wave * 16 + l16) * PT;
    const int c0a = hi16 ? 8 : 0, cb = hi16 ? 8 : 0;
    bf16x16 a0 = ldsA_frag(arow, 0 + c0a), a1 = ldsA_frag(arow, 32 + c0a);
    bf16x16 a2 = ldsA_frag(arow, 64 + c0a), a3 = ldsA_frag(arow, 96 + c0a);
#pragma unroll
    for (int nt = 0; nt < 8; ++nt) {
      f32x8 acc = {};
      const bf16* brow = A + (nt * 16 + l16) * PT;
      acc = wmma_bf16(a0, ldsB_frag(brow, 0 + cb), acc);
      acc = wmma_bf16(a1, ldsB_frag(brow, 32 + cb), acc);
      acc = wmma_bf16(a2, ldsB_frag(brow, 64 + cb), acc);
      acc = wmma_bf16(a3, ldsB_frag(brow, 96 + cb), acc);
      const int q = nt * 16 + l16;
#pragma unroll
      for (int rr = 0; rr < 8; ++rr) {
        int c = wave * 16 + rr + hi16 * 8;
        OT[q * PT + c] = (bf16)acc[rr];
      }
    }
  }
  __syncthreads();

  {  // post-conv + BN + shortcut + ReLU
    const int wI = VERT ? 5 : 3;
    const float* Wrow = W7 + wI * 16384 + (wave * 16 + l16) * 128;
    const int c0a = hi16 ? 8 : 0, cb = hi16 ? 8 : 0;
    bf16x16 a0 = gldA_f32(Wrow, 0 + c0a), a1 = gldA_f32(Wrow, 32 + c0a);
    bf16x16 a2 = gldA_f32(Wrow, 64 + c0a), a3 = gldA_f32(Wrow, 96 + c0a);
    float sc8[8], sh8[8];
#pragma unroll
    for (int rr = 0; rr < 8; ++rr) {
      int o = wave * 16 + rr + hi16 * 8;
      sc8[rr] = scale[wI * 128 + o];
      sh8[rr] = shift[wI * 128 + o];
    }
#pragma unroll
    for (int nt = 0; nt < 8; ++nt) {
      f32x8 acc = {};
      const bf16* brow = OT + (nt * 16 + l16) * PT;
      acc = wmma_bf16(a0, ldsB_frag(brow, 0 + cb), acc);
      acc = wmma_bf16(a1, ldsB_frag(brow, 32 + cb), acc);
      acc = wmma_bf16(a2, ldsB_frag(brow, 64 + cb), acc);
      acc = wmma_bf16(a3, ldsB_frag(brow, 96 + cb), acc);
      const int q = nt * 16 + l16;
#pragma unroll
      for (int rr = 0; rr < 8; ++rr) {
        int o = wave * 16 + rr + hi16 * 8;
        float v = acc[rr] * sc8[rr] + sh8[rr];
        long long si = VERT ? ((((long long)b * 128 + o) * 128 + q) * 128 + r)
                            : ((((long long)b * 128 + o) * 128 + r) * 128 + q);
        v += (float)sc_in[si];
        FH[q * 128 + o] = (bf16)fmaxf(v, 0.f);
      }
    }
  }
  __syncthreads();

  const unsigned* s32 = (const unsigned*)FH;
  if (!VERT) {
    unsigned* d32 = (unsigned*)(fout + (((long long)b * 128 + r) * 128) * 128);
    for (int idx = tid; idx < 8192; idx += 256) d32[idx] = s32[idx];
  } else {
    for (int idx = tid; idx < 8192; idx += 256) {
      int q = idx >> 6, o2 = idx & 63;
      *(unsigned*)(fout + (((long long)b * 128 + q) * 128 + r) * 128 + o2 * 2) = s32[idx];
    }
  }
}

// ---- Kernel 4: fuse concat(f_h,f_v) -> 256->128 conv + BN + ReLU -----------
__global__ void __launch_bounds__(256) fuse_kernel(
    const float* __restrict__ W8, const float* __restrict__ scale,
    const float* __restrict__ shift, const bf16* __restrict__ fh,
    const bf16* __restrict__ fv, float* __restrict__ out) {
  extern __shared__ char smem[];
  bf16* FHt = (bf16*)smem;                    // [w][c]
  bf16* FVt = (bf16*)(smem + 128 * PT * 2);

  const int b = blockIdx.x >> 7, h = blockIdx.x & 127;
  const int tid = threadIdx.x, lane = tid & 31, wave = tid >> 5;
  const int hi16 = (lane >> 4) & 1, l16 = lane & 15;

  const long long base = (((long long)b * 128 + h) * 128) * 128;
  for (int idx = tid; idx < 2048; idx += 256) {
    int q = idx >> 4, seg = (idx & 15) * 8;
    async_ld_b128(fh + base + q * 128 + seg, FHt + q * PT + seg);
    async_ld_b128(fv + base + q * 128 + seg, FVt + q * PT + seg);
  }
  wait_async0();
  __syncthreads();

  const float* Wrow = W8 + (wave * 16 + l16) * 256;
  const int c0a = hi16 ? 8 : 0, cb = hi16 ? 8 : 0;
  bf16x16 a[8];
#pragma unroll
  for (int cc = 0; cc < 8; ++cc) a[cc] = gldA_f32(Wrow, cc * 32 + c0a);
  float sc8[8], sh8[8];
#pragma unroll
  for (int rr = 0; rr < 8; ++rr) {
    int o = wave * 16 + rr + hi16 * 8;
    sc8[rr] = scale[7 * 128 + o];
    sh8[rr] = shift[7 * 128 + o];
  }
#pragma unroll
  for (int nt = 0; nt < 8; ++nt) {
    f32x8 acc = {};
    const bf16* bh = FHt + (nt * 16 + l16) * PT;
    const bf16* bv = FVt + (nt * 16 + l16) * PT;
#pragma unroll
    for (int cc = 0; cc < 4; ++cc) acc = wmma_bf16(a[cc], ldsB_frag(bh, cc * 32 + cb), acc);
#pragma unroll
    for (int cc = 0; cc < 4; ++cc) acc = wmma_bf16(a[4 + cc], ldsB_frag(bv, cc * 32 + cb), acc);
    const int w = nt * 16 + l16;
#pragma unroll
    for (int rr = 0; rr < 8; ++rr) {
      int o = wave * 16 + rr + hi16 * 8;
      out[(((long long)b * 128 + o) * 128 + h) * 128 + w] =
          fmaxf(acc[rr] * sc8[rr] + sh8[rr], 0.f);
    }
  }
}

// ---- Host launch ------------------------------------------------------------
extern "C" void kernel_launch(void* const* d_in, const int* in_sizes, int n_in,
                              void* d_out, int out_size, void* d_ws, size_t ws_size,
                              hipStream_t stream) {
  const float* x     = (const float*)d_in[0];
  const float* W7    = (const float*)d_in[1];
  const float* W8    = (const float*)d_in[2];
  const float* scale = (const float*)d_in[3];
  const float* shift = (const float*)d_in[4];
  float* out = (float*)d_out;

  bf16* ws    = (bf16*)d_ws;
  bf16* theta = ws;
  bf16* phi   = ws + NT;
  bf16* g     = ws + 2 * NT;
  bf16* sch   = ws + 3 * NT;
  bf16* scv   = ws + 4 * NT;
  bf16* fh    = ws + 5 * NT;
  bf16* fv    = ws + 6 * NT;

  (void)hipFuncSetAttribute((const void*)proj_kernel,
                            hipFuncAttributeMaxDynamicSharedMemorySize, (int)SMEM_PROJ);
  (void)hipFuncSetAttribute((const void*)attn_kernel<0>,
                            hipFuncAttributeMaxDynamicSharedMemorySize, (int)SMEM_ATTN);
  (void)hipFuncSetAttribute((const void*)attn_kernel<1>,
                            hipFuncAttributeMaxDynamicSharedMemorySize, (int)SMEM_ATTN);
  (void)hipFuncSetAttribute((const void*)fuse_kernel,
                            hipFuncAttributeMaxDynamicSharedMemorySize, (int)SMEM_FUSE);

  dim3 grid(2048), block(256);
  proj_kernel<<<grid, block, SMEM_PROJ, stream>>>(x, W7, scale, shift, theta, phi, g,
                                                  sch, scv);
  attn_kernel<0><<<grid, block, SMEM_ATTN, stream>>>(W7, scale, shift, theta, phi, g,
                                                     sch, fh);
  attn_kernel<1><<<grid, block, SMEM_ATTN, stream>>>(W7, scale, shift, theta, phi, g,
                                                     scv, fv);
  fuse_kernel<<<grid, block, SMEM_FUSE, stream>>>(W8, scale, shift, fh, fv, out);
}
